// MoE_60404420051198
// MI455X (gfx1250) — matrix operand order repaired and checked by reading, not compiled, not run on hardware
//
#include <hip/hip_runtime.h>

// ---------------------------------------------------------------------------
// MoE (top-2 of 8 experts), D=1024, H=2048, T=2048, fp32 in/out.
//
// Fast path (needs ~84 MB workspace):
//   1) router: sigmoid scores + top-2 -> wt[t][e]
//   2) per-expert token compaction (list entry = t*2+slot = partial row id)
//   3) gathered GEMM, one workgroup per (expert, 32-token tile):
//        h = wt^2 * relu(x @ wu^T)^2 ; partial[t*2+slot] = h @ wd^T
//      bf16 WMMA 16x16x32 for both GEMMs; B-fragments reused across 2 M-tiles
//      (halves L2 weight traffic vs M=16); weights pre-converted to bf16 so
//      B-frags are direct b128 loads; x rows gathered into LDS with
//      GLOBAL_LOAD_ASYNC_TO_LDS_B128 (ASYNCcnt).
//   4) deterministic reduce: out[t] = partial[t*2] + partial[t*2+1]
//
// Fallback (tiny workspace): fused dense-skip M=16 kernel, fp32 weights.
// ---------------------------------------------------------------------------

#define DIM   1024
#define NEXP  8
#define HID   2048
#define TOK   2048
#define XP    (DIM + 8)       // LDS pitch (bf16); %128==8 -> 4-bank row rotate
#define HP    (HID + 8)

typedef __attribute__((ext_vector_type(16))) __bf16 v16bf;
typedef __attribute__((ext_vector_type(8)))  float  v8f;

union Frag { v16bf v; unsigned int u[8]; uint4 q[2]; };

__device__ __forceinline__ unsigned int pack2_bf16(float a, float b) {
  union { __bf16 h[2]; unsigned int u; } r;
  r.h[0] = (__bf16)a;
  r.h[1] = (__bf16)b;
  return r.u;
}

__device__ __forceinline__ unsigned short cvt_bf16(float a) {
  union { __bf16 h; unsigned short u; } r;
  r.h = (__bf16)a;
  return r.u;
}

// ---------------------------------------------------------------------------
// Router
// ---------------------------------------------------------------------------
__global__ void moe_router(const float* __restrict__ x,
                           const float* __restrict__ gate_w,
                           const float* __restrict__ bias,
                           float* __restrict__ wt) {
  __shared__ float part[NEXP][32];
  __shared__ float score[NEXP];
  const int t    = blockIdx.x;
  const int lane = threadIdx.x & 31;
  const int e    = threadIdx.x >> 5;

  const float* xr = x + (size_t)t * DIM;
  const float* gr = gate_w + (size_t)e * DIM;
  float s = 0.f;
  for (int k = lane; k < DIM; k += 32) s += xr[k] * gr[k];
  part[e][lane] = s;
  __syncthreads();

  if (threadIdx.x < NEXP) {
    float acc = 0.f;
    #pragma unroll
    for (int i = 0; i < 32; ++i) acc += part[threadIdx.x][i];
    score[threadIdx.x] = 1.0f / (1.0f + __expf(-acc));
  }
  __syncthreads();

  if (threadIdx.x == 0) {
    float biased[NEXP];
    #pragma unroll
    for (int k = 0; k < NEXP; ++k) biased[k] = score[k] + bias[k];
    int i0 = 0; float m0 = biased[0];
    #pragma unroll
    for (int k = 1; k < NEXP; ++k) if (biased[k] > m0) { m0 = biased[k]; i0 = k; }
    int i1 = -1; float m1 = -3.4e38f;
    #pragma unroll
    for (int k = 0; k < NEXP; ++k)
      if (k != i0 && biased[k] > m1) { m1 = biased[k]; i1 = k; }
    #pragma unroll
    for (int k = 0; k < NEXP; ++k)
      wt[(size_t)t * NEXP + k] = (k == i0 || k == i1) ? score[k] : 0.0f;
  }
}

// ---------------------------------------------------------------------------
// Token compaction: per expert, list of (token*2 + slot). slot = 0/1 in
// expert-index order; entry value doubles as the partial-buffer row index.
// ---------------------------------------------------------------------------
__global__ void zero_cnt(int* cnt) { if (threadIdx.x < NEXP) cnt[threadIdx.x] = 0; }

__global__ void build_lists(const float* __restrict__ wt,
                            int* __restrict__ cnt, int* __restrict__ lists) {
  const int t = blockIdx.x * blockDim.x + threadIdx.x;
  if (t >= TOK) return;
  int slot = 0;
  #pragma unroll
  for (int e = 0; e < NEXP; ++e) {
    const float w = wt[(size_t)t * NEXP + e];
    if (w != 0.0f) {
      const int pos = atomicAdd(&cnt[e], 1);
      lists[e * TOK + pos] = (t << 1) | slot;
      slot++;
    }
  }
}

// ---------------------------------------------------------------------------
// fp32 -> bf16 bulk weight conversion (4 elems / thread)
// ---------------------------------------------------------------------------
__global__ void cvt_f32_bf16(const float* __restrict__ src,
                             unsigned short* __restrict__ dst, int n4) {
  const int i = blockIdx.x * blockDim.x + threadIdx.x;
  if (i >= n4) return;
  const float4 v = ((const float4*)src)[i];
  uint2 p;
  p.x = pack2_bf16(v.x, v.y);
  p.y = pack2_bf16(v.z, v.w);
  ((uint2*)dst)[i] = p;
}

// ---------------------------------------------------------------------------
// B-fragment loader. B (32x16 bf16): lane n=L&15; vgpr j -> K=(L<16?0:16)+2j
// = 16 consecutive K elements of one weight row -> two uint4 loads.
// ---------------------------------------------------------------------------
template <bool PRE>
__device__ __forceinline__ void load_b(Frag& B, const float* wf,
                                       const unsigned short* wb,
                                       size_t rowoff, int k0) {
  if constexpr (PRE) {
    const unsigned short* p = wb + rowoff + k0;
    B.q[0] = *(const uint4*)(p);
    B.q[1] = *(const uint4*)(p + 8);
  } else {
    const float* p = wf + rowoff + k0;
    #pragma unroll
    for (int j = 0; j < 8; ++j) {
      const float2 w2 = *(const float2*)(p + 2 * j);
      B.u[j] = pack2_bf16(w2.x, w2.y);
    }
  }
}

__device__ __forceinline__ void load_a(Frag& A, const unsigned short* tile,
                                       int pitch, int row, int ks, int aKb) {
  const unsigned short* arow = tile + (size_t)row * pitch + ks + aKb;
  A.q[0] = *(const uint4*)(arow);
  A.q[1] = *(const uint4*)(arow + 8);
}

// ===========================================================================
// Fast path: gathered GEMM, one workgroup = (expert e, tile of 32 tokens).
// LDS: info (512 B) | xa 32 x XP bf16 (66048) | hh 32 x HP bf16 (131584)
// ===========================================================================
#define G_INFO_OFF 0
#define G_XA_OFF   512
#define G_HH_OFF   (G_XA_OFF + 32 * XP * 2)     // 66560
#define G_LDS      (G_HH_OFF + 32 * HP * 2)     // 198144 (< 320 KB WGP)

__global__ void __launch_bounds__(256, 1)
moe_gemm_gather(const float* __restrict__ x,
                const unsigned short* __restrict__ w_up_b,
                const unsigned short* __restrict__ w_down_b,
                const float* __restrict__ wt,
                const int* __restrict__ cnt,
                const int* __restrict__ lists,
                float* __restrict__ partial) {
  extern __shared__ unsigned char smem[];
  int*            lst = (int*)(smem + G_INFO_OFF);          // 32 entries
  float*          lwt = (float*)(smem + G_INFO_OFF + 128);  // 32 weights
  unsigned short* xa  = (unsigned short*)(smem + G_XA_OFF);
  unsigned short* hh  = (unsigned short*)(smem + G_HH_OFF);

  const int e    = blockIdx.x >> 6;        // 64 tiles per expert
  const int base = (blockIdx.x & 63) * 32;
  const int cnte = cnt[e];
  if (base >= cnte) return;                // uniform early exit

  const int tid  = threadIdx.x;
  const int lane = tid & 31;
  const int wv   = tid >> 5;               // 8 waves
  const int ln   = lane & 15;
  const int hi   = lane >> 4;
  const int aKb  = hi ? 16 : 0;
  const int bKb  = hi ? 16 : 0;
  const int mb   = hi * 8;

  if (tid < 32) {
    const int idx = base + tid;
    int v = -1; float w = 0.0f;
    if (idx < cnte) {
      v = lists[e * TOK + idx];
      w = wt[(size_t)(v >> 1) * NEXP + e];
    }
    lst[tid] = v;
    lwt[tid] = w;
  }
  __syncthreads();

  // ---- gather 32 token rows (fp32) into hh staging via async-to-LDS b128;
  //      pad rows zero-filled.
  {
    const unsigned hhbase = (unsigned)(uintptr_t)hh;
    const unsigned long long xb = (unsigned long long)(uintptr_t)x;
    for (int i = 0; i < 32; ++i) {
      const int v = lst[i];
      const unsigned lo = hhbase + (unsigned)(i * 4096) + (unsigned)tid * 16u;
      if (v >= 0) {
        const unsigned long long ga =
            xb + (unsigned long long)(unsigned)(v >> 1) * (DIM * 4ull) +
            (unsigned long long)tid * 16u;
        asm volatile("global_load_async_to_lds_b128 %0, %1, off"
                     :: "v"(lo), "v"(ga) : "memory");
      } else {
        uint4 z; z.x = z.y = z.z = z.w = 0u;
        *(uint4*)((unsigned char*)hh + (size_t)i * 4096 + (size_t)tid * 16) = z;
      }
    }
    asm volatile("s_wait_asynccnt 0x0" ::: "memory");
    __syncthreads();
  }

  // ---- convert staged fp32 rows -> padded bf16 A tile (unscaled)
  {
    const int m  = tid >> 3;
    const int cb = (tid & 7) * 128;
    const float4* src = (const float4*)((const float*)hh + (size_t)m * DIM + cb);
    uint2* dst = (uint2*)(xa + (size_t)m * XP + cb);
    #pragma unroll
    for (int i = 0; i < 32; ++i) {
      const float4 v = src[i];
      uint2 p;
      p.x = pack2_bf16(v.x, v.y);
      p.y = pack2_bf16(v.z, v.w);
      dst[i] = p;
    }
  }

  // per-lane routing info for this lane's C rows (2 M-tiles x 8 rows)
  float w2[2][8];
  int   ts[2][8];
  #pragma unroll
  for (int mt = 0; mt < 2; ++mt)
    #pragma unroll
    for (int j = 0; j < 8; ++j) {
      const int row = mt * 16 + mb + j;
      const float s = lwt[row];
      w2[mt][j] = s * s;                  // relu(wt*z)^2 = wt^2*relu(z)^2
      ts[mt][j] = lst[row];               // partial row id, -1 for pads
    }
  __syncthreads();

  // ---- phase 1: h = wt^2 * relu(x @ wu^T)^2 ; wave owns n in [wv*256,+256)
  {
    const unsigned short* wub = w_up_b + (size_t)e * HID * DIM;
    for (int nt0 = 0; nt0 < 16; nt0 += 4) {
      const int nb = wv * 256 + nt0 * 16;
      v8f acc[2][4];
      size_t rrow[4];
      #pragma unroll
      for (int u = 0; u < 4; ++u) {
        acc[0][u] = (v8f){};
        acc[1][u] = (v8f){};
        rrow[u] = (size_t)(nb + u * 16 + ln) * DIM;
      }
      for (int ks = 0; ks < DIM; ks += 32) {
        Frag A0, A1;
        load_a(A0, xa, XP, ln,      ks, aKb);
        load_a(A1, xa, XP, 16 + ln, ks, aKb);
        #pragma unroll
        for (int u = 0; u < 4; ++u) {
          Frag B;
          load_b<true>(B, nullptr, wub, rrow[u], ks + bKb);
          acc[0][u] = __builtin_amdgcn_wmma_f32_16x16x32_bf16(
              false, A0.v, false, B.v, (short)0, acc[0][u], false, false);
          acc[1][u] = __builtin_amdgcn_wmma_f32_16x16x32_bf16(
              false, A1.v, false, B.v, (short)0, acc[1][u], false, false);
        }
      }
      #pragma unroll
      for (int mt = 0; mt < 2; ++mt)
        #pragma unroll
        for (int u = 0; u < 4; ++u) {
          const int n = nb + u * 16 + ln;
          #pragma unroll
          for (int j = 0; j < 8; ++j) {
            const float r = fmaxf(acc[mt][u][j], 0.0f);
            hh[(size_t)(mt * 16 + mb + j) * HP + n] = cvt_bf16(w2[mt][j] * r * r);
          }
        }
    }
  }
  __syncthreads();

  // ---- phase 2: partial[t*2+slot] = h @ wd^T ; wave owns d in [wv*128,+128)
  {
    const unsigned short* wdb = w_down_b + (size_t)e * DIM * HID;
    #pragma unroll
    for (int c = 0; c < 2; ++c) {
      const int db = wv * 128 + c * 64;
      v8f acc[2][4];
      size_t rrow[4];
      #pragma unroll
      for (int u = 0; u < 4; ++u) {
        acc[0][u] = (v8f){};
        acc[1][u] = (v8f){};
        rrow[u] = (size_t)(db + u * 16 + ln) * HID;
      }
      for (int ks = 0; ks < HID; ks += 32) {
        Frag A0, A1;
        load_a(A0, hh, HP, ln,      ks, aKb);
        load_a(A1, hh, HP, 16 + ln, ks, aKb);
        #pragma unroll
        for (int u = 0; u < 4; ++u) {
          Frag B;
          load_b<true>(B, nullptr, wdb, rrow[u], ks + bKb);
          acc[0][u] = __builtin_amdgcn_wmma_f32_16x16x32_bf16(
              false, A0.v, false, B.v, (short)0, acc[0][u], false, false);
          acc[1][u] = __builtin_amdgcn_wmma_f32_16x16x32_bf16(
              false, A1.v, false, B.v, (short)0, acc[1][u], false, false);
        }
      }
      #pragma unroll
      for (int mt = 0; mt < 2; ++mt)
        #pragma unroll
        for (int u = 0; u < 4; ++u) {
          const int d = db + u * 16 + ln;
          #pragma unroll
          for (int j = 0; j < 8; ++j)
            if (ts[mt][j] >= 0)
              partial[(size_t)ts[mt][j] * DIM + d] = acc[mt][u][j];
        }
    }
  }
}

// ---------------------------------------------------------------------------
// Deterministic combine: out[t] = partial[t*2] + partial[t*2+1]
// ---------------------------------------------------------------------------
__global__ void moe_reduce(const float* __restrict__ p, float* __restrict__ out) {
  const int i = blockIdx.x * blockDim.x + threadIdx.x;   // one float4 of out
  const int t = i >> 8;                                   // DIM/4 = 256 per row
  const int q = i & 255;
  const float4 a = ((const float4*)p)[(size_t)(t * 2)     * 256 + q];
  const float4 b = ((const float4*)p)[(size_t)(t * 2 + 1) * 256 + q];
  float4 r;
  r.x = a.x + b.x; r.y = a.y + b.y; r.z = a.z + b.z; r.w = a.w + b.w;
  ((float4*)out)[i] = r;
}

// ===========================================================================
// Fallback: fused dense-skip kernel (M=16, fp32 weights on the fly) for
// the case where the workspace cannot hold bf16 weights + partials.
// ===========================================================================
#define MT        16
#define WTS_OFF   0
#define FLAG_OFF  64
#define XA_OFF    80
#define HH_OFF    (XA_OFF + MT * XP * 2)
#define LDS_BYTES (HH_OFF + MT * HP * 2)

#define ALOAD(OFF)                                                      \
  asm volatile("global_load_async_to_lds_b128 %0, %1, off offset:" #OFF \
               :: "v"(lbase), "v"(gbase) : "memory")

__global__ void __launch_bounds__(256, 1)
moe_expert_dense(const float* __restrict__ x,
                 const float* __restrict__ w_up_f,
                 const float* __restrict__ w_down_f,
                 const float* __restrict__ wt,
                 float* __restrict__ out) {
  extern __shared__ unsigned char smem[];
  float*          wts  = (float*)(smem + WTS_OFF);
  int*            flag = (int*)  (smem + FLAG_OFF);
  unsigned short* xa   = (unsigned short*)(smem + XA_OFF);
  unsigned short* hh   = (unsigned short*)(smem + HH_OFF);

  const int t0   = blockIdx.x * MT;
  const int tid  = threadIdx.x;
  const int lane = tid & 31;
  const int wv   = tid >> 5;
  const int ln   = lane & 15;
  const int hi   = lane >> 4;
  const int aKb  = hi ? 16 : 0;
  const int bKb  = hi ? 16 : 0;
  const int mb   = hi * 8;

  {
    const unsigned lbase = (unsigned)(uintptr_t)hh + (unsigned)tid * 16u;
    const unsigned long long gbase =
        (unsigned long long)(uintptr_t)(x + (size_t)t0 * DIM) +
        (unsigned long long)tid * 16u;
    ALOAD(0);     ALOAD(4096);  ALOAD(8192);  ALOAD(12288);
    ALOAD(16384); ALOAD(20480); ALOAD(24576); ALOAD(28672);
    ALOAD(32768); ALOAD(36864); ALOAD(40960); ALOAD(45056);
    ALOAD(49152); ALOAD(53248); ALOAD(57344); ALOAD(61440);
    asm volatile("s_wait_asynccnt 0x0" ::: "memory");
    __syncthreads();

    const int m  = tid >> 4;
    const int kb = (tid & 15) * 64;
    const float4* src = (const float4*)((const float*)hh + (size_t)m * DIM + kb);
    uint2* dst = (uint2*)(xa + (size_t)m * XP + kb);
    #pragma unroll
    for (int i = 0; i < 16; ++i) {
      const float4 v = src[i];
      uint2 p;
      p.x = pack2_bf16(v.x, v.y);
      p.y = pack2_bf16(v.z, v.w);
      dst[i] = p;
    }
  }

  v8f accO[8];
  #pragma unroll
  for (int i = 0; i < 8; ++i) accO[i] = (v8f){};

  for (int e = 0; e < NEXP; ++e) {
    __syncthreads();
    if (tid < MT) wts[tid] = wt[(size_t)(t0 + tid) * NEXP + e];
    if (tid == MT) flag[0] = 0;
    __syncthreads();
    if (tid < MT && wts[tid] != 0.0f) atomicOr(flag, 1);
    __syncthreads();
    if (flag[0] == 0) continue;

    float w2[8];
    #pragma unroll
    for (int j = 0; j < 8; ++j) { const float s = wts[mb + j]; w2[j] = s * s; }

    {
      const float* wuf = w_up_f + (size_t)e * HID * DIM;
      for (int nt0 = 0; nt0 < 16; nt0 += 4) {
        const int nb = wv * 256 + nt0 * 16;
        v8f acc[4];
        size_t rrow[4];
        #pragma unroll
        for (int u = 0; u < 4; ++u) {
          acc[u] = (v8f){};
          rrow[u] = (size_t)(nb + u * 16 + ln) * DIM;
        }
        for (int ks = 0; ks < DIM; ks += 32) {
          Frag A;
          load_a(A, xa, XP, ln, ks, aKb);
          #pragma unroll
          for (int u = 0; u < 4; ++u) {
            Frag B;
            load_b<false>(B, wuf, nullptr, rrow[u], ks + bKb);
            acc[u] = __builtin_amdgcn_wmma_f32_16x16x32_bf16(
                false, A.v, false, B.v, (short)0, acc[u], false, false);
          }
        }
        #pragma unroll
        for (int u = 0; u < 4; ++u) {
          const int n = nb + u * 16 + ln;
          #pragma unroll
          for (int j = 0; j < 8; ++j) {
            const float r = fmaxf(acc[u][j], 0.0f);
            hh[(size_t)(mb + j) * HP + n] = cvt_bf16(w2[j] * r * r);
          }
        }
      }
    }
    __syncthreads();

    {
      const float* wdf = w_down_f + (size_t)e * DIM * HID;
      #pragma unroll
      for (int c = 0; c < 2; ++c) {
        const int db = wv * 128 + c * 64;
        size_t rrow[4];
        #pragma unroll
        for (int u = 0; u < 4; ++u)
          rrow[u] = (size_t)(db + u * 16 + ln) * HID;
        for (int ks = 0; ks < HID; ks += 32) {
          Frag A;
          load_a(A, hh, HP, ln, ks, aKb);
          #pragma unroll
          for (int u = 0; u < 4; ++u) {
            Frag B;
            load_b<false>(B, wdf, nullptr, rrow[u], ks + bKb);
            accO[c * 4 + u] = __builtin_amdgcn_wmma_f32_16x16x32_bf16(
                false, A.v, false, B.v, (short)0, accO[c * 4 + u], false, false);
          }
        }
      }
    }
  }

  __syncthreads();
  {
    float* of = (float*)hh;
    #pragma unroll
    for (int c = 0; c < 2; ++c)
      #pragma unroll
      for (int u = 0; u < 4; ++u) {
        const int d = wv * 128 + c * 64 + u * 16 + ln;
        #pragma unroll
        for (int j = 0; j < 8; ++j)
          of[(size_t)(mb + j) * DIM + d] = accO[c * 4 + u][j];
      }
  }
  __syncthreads();
  {
    const int m  = tid >> 4;
    const int kb = (tid & 15) * 64;
    const float4* src = (const float4*)((const float*)hh + (size_t)m * DIM + kb);
    float4* dst = (float4*)(out + (size_t)(t0 + m) * DIM + kb);
    #pragma unroll
    for (int i = 0; i < 16; ++i) dst[i] = src[i];
  }
}

// ---------------------------------------------------------------------------
extern "C" void kernel_launch(void* const* d_in, const int* in_sizes, int n_in,
                              void* d_out, int out_size, void* d_ws, size_t ws_size,
                              hipStream_t stream) {
  const float* x           = (const float*)d_in[0];   // (1, 2048, 1024)
  const float* gate_w      = (const float*)d_in[1];   // (8, 1024)
  const float* w_up        = (const float*)d_in[2];   // (8, 2048, 1024)
  const float* w_down      = (const float*)d_in[3];   // (8, 1024, 2048)
  const float* expert_bias = (const float*)d_in[4];   // (8,)
  float* out = (float*)d_out;                         // (1, 2048, 1024)

  // workspace layout
  const size_t wt_bytes   = (size_t)TOK * NEXP * 4;          // 64 KB
  const size_t cnt_bytes  = 128;
  const size_t list_bytes = (size_t)NEXP * TOK * 4;          // 64 KB
  const size_t part_bytes = (size_t)TOK * 2 * DIM * 4;       // 16 MB
  const size_t w_elems    = (size_t)NEXP * HID * DIM;        // 16.7M / tensor
  const size_t need = wt_bytes + cnt_bytes + list_bytes + part_bytes
                    + 2 * w_elems * 2;                       // ~84 MB

  float* wtb   = (float*)d_ws;
  int*   cntb  = (int*)((char*)d_ws + wt_bytes);
  int*   lists = (int*)((char*)d_ws + wt_bytes + cnt_bytes);
  float* part  = (float*)((char*)d_ws + wt_bytes + cnt_bytes + list_bytes);
  unsigned short* wu_b =
      (unsigned short*)((char*)d_ws + wt_bytes + cnt_bytes + list_bytes + part_bytes);
  unsigned short* wd_b = wu_b + w_elems;

  moe_router<<<TOK, 256, 0, stream>>>(x, gate_w, expert_bias, wtb);

  if (ws_size >= need) {
    zero_cnt<<<1, 32, 0, stream>>>(cntb);
    build_lists<<<TOK / 256, 256, 0, stream>>>(wtb, cntb, lists);
    const int n4 = (int)(w_elems / 4);
    cvt_f32_bf16<<<(n4 + 255) / 256, 256, 0, stream>>>(w_up,   wu_b, n4);
    cvt_f32_bf16<<<(n4 + 255) / 256, 256, 0, stream>>>(w_down, wd_b, n4);
    moe_gemm_gather<<<NEXP * 64, 256, G_LDS, stream>>>(
        x, wu_b, wd_b, wtb, cntb, lists, part);
    moe_reduce<<<(TOK * (DIM / 4)) / 256, 256, 0, stream>>>(part, out);
  } else {
    moe_expert_dense<<<TOK / MT, 256, LDS_BYTES, stream>>>(
        x, w_up, w_down, wtb, out);
  }
}
